// DMPNNEncoder_73443940762169
// MI455X (gfx1250) — matrix operand order repaired and checked by reading, not compiled, not run on hardware
//
#include <hip/hip_runtime.h>
#include <hip/hip_bf16.h>
#include <stdint.h>

// ---------------------------------------------------------------------------
// D-MPNN encoder for MI455X (gfx1250), wave32 + WMMA f32_16x16x32_f16.
// Exact tiling: 800000 edges = 50000x16, 50000 nodes = 3125x16 -> no guards.
// ---------------------------------------------------------------------------

#define N_NODES 50000
#define N_EDGES 800000
#define NODE_F  133
#define EDGE_F  14
#define HID     128
#define NUM_GRAPHS 512

typedef _Float16 half_t;
typedef __attribute__((ext_vector_type(16))) _Float16 v16h;
typedef __attribute__((ext_vector_type(8)))  _Float16 v8h;
typedef __attribute__((ext_vector_type(4)))  _Float16 v4h;
typedef __attribute__((ext_vector_type(8)))  float    v8f;
typedef __attribute__((ext_vector_type(4)))  float    v4f;
typedef __attribute__((ext_vector_type(4)))  int      v4i;

// async-to-LDS builtin parameter types (per clang: "vV4i*1V4i*3IiIi")
typedef __attribute__((address_space(1))) v4i* gas_v4i_p;
typedef __attribute__((address_space(3))) v4i* las_v4i_p;

static __device__ __forceinline__ v8f wmma16(v16h a, v16h b, v8f c) {
  return __builtin_amdgcn_wmma_f32_16x16x32_f16(
      false, a, false, b, (short)0, c, false, false);
}

static __device__ __forceinline__ v16h pack16(v8h lo, v8h hi) {
  v16h r;
#pragma unroll
  for (int i = 0; i < 8; ++i) { r[i] = lo[i]; r[i + 8] = hi[i]; }
  return r;
}

// Fill LDS from a global f16 buffer; async-to-LDS path when available.
static __device__ __forceinline__ void lds_fill(half_t* dstLds,
                                                const half_t* srcGlobal,
                                                int totalHalves,
                                                int tid, int nthreads) {
#if __has_builtin(__builtin_amdgcn_global_load_async_to_lds_b128)
  for (int i = tid; i < totalHalves / 8; i += nthreads) {
    __builtin_amdgcn_global_load_async_to_lds_b128(
        (gas_v4i_p)(uintptr_t)&srcGlobal[i * 8],
        (las_v4i_p)&dstLds[i * 8],
        0, 0);
  }
  asm volatile("s_wait_asynccnt 0x0" ::: "memory");
#else
  for (int i = tid; i < totalHalves / 8; i += nthreads)
    *(v8h*)&dstLds[i * 8] = *(const v8h*)&srcGlobal[i * 8];
#endif
}

// ---------------------------------------------------------------------------
// Weight conversion kernels (once per call, tiny): f32 -> f16 transposed.
// ---------------------------------------------------------------------------
__global__ void k_w1t(const float* __restrict__ W, half_t* __restrict__ Wt) {
  const int t = blockIdx.x * 256 + threadIdx.x;          // 128*160
  if (t >= HID * 160) return;
  const int n = t / 160, k = t - n * 160;
  Wt[t] = (half_t)((k < NODE_F + EDGE_F) ? W[k * HID + n] : 0.0f);
}
__global__ void k_w2t(const float* __restrict__ W, half_t* __restrict__ Wt) {
  const int t = blockIdx.x * 256 + threadIdx.x;          // 128*128
  if (t >= HID * HID) return;
  const int n = t >> 7, k = t & 127;
  Wt[t] = (half_t)W[k * HID + n];
}
__global__ void k_w3t(const float* __restrict__ W, half_t* __restrict__ Wt) {
  const int t = blockIdx.x * 256 + threadIdx.x;          // 128*288
  if (t >= HID * 288) return;
  const int n = t / 288, k = t - n * 288;
  Wt[t] = (half_t)((k < NODE_F + HID) ? W[k * HID + n] : 0.0f);
}

// ---------------------------------------------------------------------------
// Kernel 1: h0 = relu(concat(x[src], edge_attr) @ W1)   (E x 147)@(147 x 128)
// K padded 147 -> 160 (5 k-steps). One 16-row tile per wave, 4 waves/block.
// ---------------------------------------------------------------------------
__global__ __launch_bounds__(128) void k_h0(
    const float* __restrict__ x, const int* __restrict__ ei,
    const float* __restrict__ ea, const half_t* __restrict__ W1t,
    half_t* __restrict__ h0)
{
  __shared__ __align__(16) half_t Wt[HID][160];
  __shared__ __align__(16) half_t At[4][16][160];

  const int tid  = threadIdx.x;
  const int wave = tid >> 5;
  const int lane = tid & 31;
  const int col  = lane & 15;
  const int grp  = lane >> 4;

  lds_fill(&Wt[0][0], W1t, HID * 160, tid, 128);

  const long e0 = ((long)blockIdx.x * 4 + wave) * 16;

  // A tile: row-uniform gather of node features + edge features
  for (int r = 0; r < 16; ++r) {
    const long e = e0 + r;
    const int  s = ei[e];                     // uniform per row
    const float* xrow = &x[(long)s * NODE_F];
    for (int c = lane; c < 160; c += 32) {
      float v = 0.0f;
      if (c < NODE_F)                 v = xrow[c];
      else if (c < NODE_F + EDGE_F)   v = ea[e * EDGE_F + (c - NODE_F)];
      At[wave][r][c] = (half_t)v;
    }
  }
  __syncthreads();

  v16h a[5];
#pragma unroll
  for (int ks = 0; ks < 5; ++ks) {
    v8h lo = *(const v8h*)&At[wave][col][ks * 32 + grp * 8];
    v8h hi = *(const v8h*)&At[wave][col][ks * 32 + 16 + grp * 8];
    a[ks] = pack16(lo, hi);
  }

  half_t* hb = &At[wave][0][0];               // reuse A tile as 16x128 staging
#pragma unroll
  for (int n = 0; n < 8; ++n) {
    v8f acc = {0.f, 0.f, 0.f, 0.f, 0.f, 0.f, 0.f, 0.f};
#pragma unroll
    for (int ks = 0; ks < 5; ++ks) {
      v8h lo = *(const v8h*)&Wt[n * 16 + col][ks * 32 + grp * 8];
      v8h hi = *(const v8h*)&Wt[n * 16 + col][ks * 32 + 16 + grp * 8];
      acc = wmma16(a[ks], pack16(lo, hi), acc);
    }
#pragma unroll
    for (int j = 0; j < 8; ++j) {
      float v = acc[j];
      v = v > 0.0f ? v : 0.0f;
      hb[(j + grp * 8) * HID + n * 16 + col] = (half_t)v;
    }
  }
  // coalesced write-out: 8 x b128 per lane
#pragma unroll
  for (int i = 0; i < 8; ++i) {
    const int chunk = lane + i * 32;          // 0..255
    const int r = chunk >> 4, cc = (chunk & 15) * 8;
    *(v8h*)&h0[(e0 + r) * HID + cc] = *(const v8h*)&hb[r * HID + cc];
  }
}

// ---------------------------------------------------------------------------
__global__ void k_zero(float* __restrict__ p, long n) {
  long t = (long)blockIdx.x * 256 + threadIdx.x;
  if (t < n) p[t] = 0.0f;
}

// ---------------------------------------------------------------------------
// node_sum[dst[e]] += h[e]   (f16 h -> f32 atomic accumulate), E*16 threads
// ---------------------------------------------------------------------------
__global__ __launch_bounds__(256) void k_segsum(
    const half_t* __restrict__ h, const int* __restrict__ dst,
    float* __restrict__ ns)
{
  const long t = (long)blockIdx.x * 256 + threadIdx.x;
  const long e = t >> 4;
  const int  q = (int)(t & 15);
  const int  d = dst[e];
  const v8h hv = *(const v8h*)&h[e * HID + q * 8];
  float* base = &ns[(long)d * HID + q * 8];
#pragma unroll
  for (int i = 0; i < 8; ++i) atomicAdd(base + i, (float)hv[i]);
}

// ---------------------------------------------------------------------------
// Kernel 4: h_out = relu(h0 + (node_sum[src] - h_in[rev]) @ W2)
// ---------------------------------------------------------------------------
__global__ __launch_bounds__(128) void k_iter(
    const half_t* __restrict__ h_in, const half_t* __restrict__ h0,
    const int* __restrict__ ei, const int* __restrict__ rev,
    const float* __restrict__ ns, const half_t* __restrict__ W2t,
    half_t* __restrict__ h_out)
{
  __shared__ __align__(16) half_t Wt[HID][HID];
  __shared__ __align__(16) half_t At[4][16][HID];

  const int tid  = threadIdx.x;
  const int wave = tid >> 5;
  const int lane = tid & 31;
  const int col  = lane & 15;
  const int grp  = lane >> 4;

  lds_fill(&Wt[0][0], W2t, HID * HID, tid, 128);

  const long e0 = ((long)blockIdx.x * 4 + wave) * 16;

  // prefetch h0 tile (epilogue operand), one 128B line per lane
  __builtin_prefetch(&h0[e0 * HID + (long)lane * 64], 0, 0);

  // A tile: m[e] = node_sum[src[e]] - h_in[rev[e]], vectorized, row-uniform
  for (int r = 0; r < 16; ++r) {
    const long e  = e0 + r;
    const int  s  = ei[e];                     // uniform per row
    const int  re = rev[e];
    const v4f nv = *(const v4f*)&ns[(long)s * HID + lane * 4];
    const v4h hv = *(const v4h*)&h_in[(long)re * HID + lane * 4];
    v4h o;
#pragma unroll
    for (int i = 0; i < 4; ++i) o[i] = (half_t)(nv[i] - (float)hv[i]);
    *(v4h*)&At[wave][r][lane * 4] = o;
  }
  __syncthreads();

  v16h a[4];
#pragma unroll
  for (int ks = 0; ks < 4; ++ks) {
    v8h lo = *(const v8h*)&At[wave][col][ks * 32 + grp * 8];
    v8h hi = *(const v8h*)&At[wave][col][ks * 32 + 16 + grp * 8];
    a[ks] = pack16(lo, hi);
  }

  half_t* hb = &At[wave][0][0];               // reuse as 16x128 staging
#pragma unroll
  for (int n = 0; n < 8; ++n) {
    v8f acc = {0.f, 0.f, 0.f, 0.f, 0.f, 0.f, 0.f, 0.f};
#pragma unroll
    for (int ks = 0; ks < 4; ++ks) {
      v8h lo = *(const v8h*)&Wt[n * 16 + col][ks * 32 + grp * 8];
      v8h hi = *(const v8h*)&Wt[n * 16 + col][ks * 32 + 16 + grp * 8];
      acc = wmma16(a[ks], pack16(lo, hi), acc);
    }
#pragma unroll
    for (int j = 0; j < 8; ++j)
      hb[(j + grp * 8) * HID + n * 16 + col] = (half_t)acc[j];
  }
  // fused h0-add + relu during coalesced write-out
#pragma unroll
  for (int i = 0; i < 8; ++i) {
    const int chunk = lane + i * 32;
    const int r = chunk >> 4, cc = (chunk & 15) * 8;
    const v8h sv = *(const v8h*)&hb[r * HID + cc];
    const v8h gv = *(const v8h*)&h0[(e0 + r) * HID + cc];
    v8h o;
#pragma unroll
    for (int k = 0; k < 8; ++k) {
      float v = (float)sv[k] + (float)gv[k];
      v = v > 0.0f ? v : 0.0f;
      o[k] = (half_t)v;
    }
    *(v8h*)&h_out[(e0 + r) * HID + cc] = o;
  }
}

// ---------------------------------------------------------------------------
// Kernel 6: node_attr = relu(concat(x, v_msg) @ W3 + b)  (N x 261)@(261 x 128)
// 5 waves/block * 625 blocks = 3125 tiles exactly. B from global f16 W3^T.
// ---------------------------------------------------------------------------
__global__ __launch_bounds__(160) void k_node(
    const float* __restrict__ x, const float* __restrict__ vmsg,
    const half_t* __restrict__ W3t, const float* __restrict__ W3b,
    float* __restrict__ na)
{
  __shared__ __align__(16) half_t At[5][16][288];

  const int tid  = threadIdx.x;
  const int wave = tid >> 5;
  const int lane = tid & 31;
  const int col  = lane & 15;
  const int grp  = lane >> 4;

  const long n0 = ((long)blockIdx.x * 5 + wave) * 16;

  for (int r = 0; r < 16; ++r) {
    const long node = n0 + r;
    for (int c = lane; c < 288; c += 32) {
      float v = 0.0f;
      if (c < NODE_F)             v = x[node * NODE_F + c];
      else if (c < NODE_F + HID)  v = vmsg[node * HID + (c - NODE_F)];
      At[wave][r][c] = (half_t)v;
    }
  }
  __syncthreads();

  v16h a[9];
#pragma unroll
  for (int ks = 0; ks < 9; ++ks) {
    v8h lo = *(const v8h*)&At[wave][col][ks * 32 + grp * 8];
    v8h hi = *(const v8h*)&At[wave][col][ks * 32 + 16 + grp * 8];
    a[ks] = pack16(lo, hi);
  }
  __syncthreads();   // fragments in regs; make LDS reuse safe vs f32 punning

  float* fb = (float*)&At[wave][0][0];        // 16x128 f32 staging (8KB<9216B)
#pragma unroll
  for (int n = 0; n < 8; ++n) {
    v8f acc = {0.f, 0.f, 0.f, 0.f, 0.f, 0.f, 0.f, 0.f};
    const long brow = (long)(n * 16 + col) * 288;
#pragma unroll
    for (int ks = 0; ks < 9; ++ks) {
      v8h lo = *(const v8h*)&W3t[brow + ks * 32 + grp * 8];
      v8h hi = *(const v8h*)&W3t[brow + ks * 32 + 16 + grp * 8];
      acc = wmma16(a[ks], pack16(lo, hi), acc);
    }
    const float bias = W3b[n * 16 + col];
#pragma unroll
    for (int j = 0; j < 8; ++j) {
      float v = acc[j] + bias;
      v = v > 0.0f ? v : 0.0f;
      fb[(j + grp * 8) * HID + n * 16 + col] = v;
    }
  }
  // coalesced f32 write-out: 16 x b128 per lane
#pragma unroll
  for (int i = 0; i < 16; ++i) {
    const int chunk = lane + i * 32;          // 0..511
    const int r = chunk >> 5, c4 = (chunk & 31) * 4;
    *(v4f*)&na[(n0 + r) * HID + c4] = *(const v4f*)&fb[r * HID + c4];
  }
}

// ---------------------------------------------------------------------------
__global__ __launch_bounds__(256) void k_pool(
    const float* __restrict__ na, const int* __restrict__ batch,
    float* __restrict__ psum, float* __restrict__ pcnt)
{
  const long t = (long)blockIdx.x * 256 + threadIdx.x;  // N*32 threads
  const long node = t >> 5;
  const int  q = (int)(t & 31);
  const int  g = batch[node];
  const v4f v = *(const v4f*)&na[node * HID + q * 4];
  float* b = &psum[(long)g * HID + q * 4];
#pragma unroll
  for (int i = 0; i < 4; ++i) atomicAdd(b + i, v[i]);
  if (q == 0) atomicAdd(&pcnt[g], 1.0f);
}

__global__ void k_finpool(const float* __restrict__ psum,
                          const float* __restrict__ pcnt,
                          float* __restrict__ pooled,
                          float* __restrict__ out_pooled)
{
  const int t = blockIdx.x * 256 + threadIdx.x;   // 512*128
  const int g = t >> 7;
  float c = pcnt[g];
  c = c > 1.0f ? c : 1.0f;
  const float v = psum[t] / c;
  pooled[t] = v;
  out_pooled[t] = v;
}

__global__ __launch_bounds__(128) void k_pred(
    const float* __restrict__ pooled, const float* __restrict__ M1w,
    const float* __restrict__ M1b, const float* __restrict__ M2w,
    const float* __restrict__ M2b, float* __restrict__ pred)
{
  __shared__ float red[HID];
  const int g = blockIdx.x, n = threadIdx.x;
  float acc = M1b[n];
  for (int k = 0; k < HID; ++k)
    acc += pooled[g * HID + k] * M1w[k * HID + n];
  acc = acc > 0.0f ? acc : 0.0f;
  red[n] = acc * M2w[n];
  __syncthreads();
  for (int s = 64; s > 0; s >>= 1) {
    if (n < s) red[n] += red[n + s];
    __syncthreads();
  }
  if (n == 0) pred[g] = red[0] + M2b[0];
}

// ---------------------------------------------------------------------------
extern "C" void kernel_launch(void* const* d_in, const int* in_sizes, int n_in,
                              void* d_out, int out_size, void* d_ws, size_t ws_size,
                              hipStream_t stream) {
  (void)in_sizes; (void)n_in; (void)out_size; (void)ws_size;

  const float* x    = (const float*)d_in[0];
  const int*   ei   = (const int*)d_in[1];      // [2][E]: src = ei, dst = ei+E
  const int*   rev  = (const int*)d_in[2];
  const float* ea   = (const float*)d_in[3];
  const int*   batch= (const int*)d_in[4];
  const float* W1   = (const float*)d_in[5];
  const float* W2   = (const float*)d_in[6];
  const float* W3w  = (const float*)d_in[7];
  const float* W3b  = (const float*)d_in[8];
  const float* M1w  = (const float*)d_in[9];
  const float* M1b  = (const float*)d_in[10];
  const float* M2w  = (const float*)d_in[11];
  const float* M2b  = (const float*)d_in[12];

  char* ws = (char*)d_ws;
  size_t off = 0;
  auto carve = [&](size_t bytes) {
    void* p = ws + off;
    off = (off + bytes + 255) & ~(size_t)255;
    return p;
  };
  const size_t SZ_H = (size_t)N_EDGES * HID * sizeof(half_t);   // 204.8 MB
  half_t* h0   = (half_t*)carve(SZ_H);
  half_t* hA   = (half_t*)carve(SZ_H);
  half_t* hB   = (half_t*)carve(SZ_H);
  float*  ns   = (float*)carve((size_t)N_NODES * HID * 4);      // node_sum/v_msg
  float*  na   = (float*)carve((size_t)N_NODES * HID * 4);
  float*  pooled = (float*)carve((size_t)NUM_GRAPHS * HID * 4);
  float*  psum   = (float*)carve((size_t)NUM_GRAPHS * HID * 4);
  float*  pcnt   = (float*)carve((size_t)NUM_GRAPHS * 4);
  half_t* W1t  = (half_t*)carve((size_t)HID * 160 * 2);
  half_t* W2t  = (half_t*)carve((size_t)HID * HID * 2);
  half_t* W3t  = (half_t*)carve((size_t)HID * 288 * 2);

  float* out_pred   = (float*)d_out;               // [512]
  float* out_pooled = (float*)d_out + NUM_GRAPHS;  // [512*128]

  const int* dst = ei + N_EDGES;

  const int edgeBlocks = N_EDGES / (16 * 4);          // 12500, exact
  const int nodeBlocks = N_NODES / (16 * 5);          // 625, exact
  const long nsElems   = (long)N_NODES * HID;         // 6.4M
  const int nsBlocks   = (int)(nsElems / 256);        // exact
  const int ssBlocks   = (int)((long)N_EDGES * 16 / 256);   // 50000, exact
  const int poolBlocks = (int)((long)N_NODES * 32 / 256);   // 6250, exact

  // f16 transposed weight copies
  k_w1t<<<(HID * 160 + 255) / 256, 256, 0, stream>>>(W1, W1t);
  k_w2t<<<(HID * HID) / 256, 256, 0, stream>>>(W2, W2t);
  k_w3t<<<(HID * 288) / 256, 256, 0, stream>>>(W3w, W3t);

  // h0 = relu(initMsg @ W1)
  k_h0<<<edgeBlocks, 128, 0, stream>>>(x, ei, ea, W1t, h0);

  // depth iteration 1: h0 -> hA
  k_zero<<<nsBlocks, 256, 0, stream>>>(ns, nsElems);
  k_segsum<<<ssBlocks, 256, 0, stream>>>(h0, dst, ns);
  k_iter<<<edgeBlocks, 128, 0, stream>>>(h0, h0, ei, rev, ns, W2t, hA);

  // depth iteration 2: hA -> hB
  k_zero<<<nsBlocks, 256, 0, stream>>>(ns, nsElems);
  k_segsum<<<ssBlocks, 256, 0, stream>>>(hA, dst, ns);
  k_iter<<<edgeBlocks, 128, 0, stream>>>(hA, h0, ei, rev, ns, W2t, hB);

  // v_msg = segment_sum(hB)
  k_zero<<<nsBlocks, 256, 0, stream>>>(ns, nsElems);
  k_segsum<<<ssBlocks, 256, 0, stream>>>(hB, dst, ns);

  // node_attr = relu([x ‖ v_msg] @ W3 + b)
  k_node<<<nodeBlocks, 160, 0, stream>>>(x, ns, W3t, W3b, na);

  // mean pool per graph
  k_zero<<<(NUM_GRAPHS * HID) / 256, 256, 0, stream>>>(psum, NUM_GRAPHS * HID);
  k_zero<<<2, 256, 0, stream>>>(pcnt, NUM_GRAPHS);
  k_pool<<<poolBlocks, 256, 0, stream>>>(na, batch, psum, pcnt);
  k_finpool<<<(NUM_GRAPHS * HID) / 256, 256, 0, stream>>>(psum, pcnt, pooled, out_pooled);

  // head MLP
  k_pred<<<NUM_GRAPHS, 128, 0, stream>>>(pooled, M1w, M1b, M2w, M2b, out_pred);
}
